// SparseLinearAttention_14431090115245
// MI455X (gfx1250) — matrix operand order repaired and checked by reading, not compiled, not run on hardware
//
#include <hip/hip_runtime.h>
#include <hip/hip_bf16.h>
#include <math.h>

// ---------------------------------------------------------------------------
// SparseLinearAttention for MI455X (gfx1250, wave32, WMMA + async-to-LDS + TDM)
// B=1, T=2048, D=512, H=8, hd=64
// ---------------------------------------------------------------------------

#define T_SEQ   2048
#define D_MODEL 512
#define H_HEADS 8
#define HD      64
#define K_CRIT  0.05f
#define K_SKIP  0.10f

// LDS tile row stride in bf16 elements (64 data + 8 pad = 144B rows,
// so consecutive rows start 36 banks apart -> conflict-free b128 fragment reads)
#define LDS_STRIDE 72

typedef unsigned short bf16_t; // raw bf16 bits in memory
typedef __attribute__((ext_vector_type(16))) __bf16 v16bf;
typedef __attribute__((ext_vector_type(8)))  float  v8f;
typedef __attribute__((ext_vector_type(4)))  unsigned int v4u;
typedef __attribute__((ext_vector_type(8)))  int v8i;
typedef __attribute__((ext_vector_type(4)))  int v4i;

union FragB { v16bf v; uint4 q[2]; };

__device__ inline bf16_t f32_to_bf16(float f) {
  unsigned u = __float_as_uint(f);
  unsigned r = (u + 0x7FFFu + ((u >> 16) & 1u)) >> 16; // round-to-nearest-even
  return (bf16_t)r;
}

// A-matrix fragment, 16x32 bf16, per ISA 7.12.2:
// lanes 0-15: row M=lane, K=0..7 (v0-3) and K=16..23 (v4-7)
// lanes 16-31: row M=lane-16, K=8..15 and K=24..31
__device__ inline v16bf load_frag_a(const bf16_t* __restrict__ base, int row,
                                    int ld, int k0, int lane) {
  const int half = lane >> 4;
  const bf16_t* p = base + (size_t)row * ld + k0;
  FragB f;
  f.q[0] = *(const uint4*)(p + half * 8);
  f.q[1] = *(const uint4*)(p + 16 + half * 8);
  return f.v;
}

// B-matrix fragment, 32x16 bf16: lane holds column n=lane&15,
// K = half*8..+7 and 16+half*8..+7 (matches A). B[k][n] = base[colrow_n*ld + k].
__device__ inline v16bf load_frag_b(const bf16_t* __restrict__ base, int colrow,
                                    int ld, int k0, int lane) {
  const int half = lane >> 4;
  const bf16_t* p = base + (size_t)colrow * ld + k0;
  FragB f;
  f.q[0] = *(const uint4*)(p + half * 8);
  f.q[1] = *(const uint4*)(p + 16 + half * 8);
  return f.v;
}

// Same fragment pulls, but sourced from LDS tiles (emit ds_load_b128).
__device__ inline v16bf lds_frag_a(const bf16_t* sbase, int row, int k0, int lane) {
  const int half = lane >> 4;
  const bf16_t* p = sbase + row * LDS_STRIDE + k0;
  FragB f;
  f.q[0] = *(const uint4*)(p + half * 8);
  f.q[1] = *(const uint4*)(p + 16 + half * 8);
  return f.v;
}

__device__ inline v16bf lds_frag_b(const bf16_t* sbase, int colrow, int k0, int lane) {
  const int half = lane >> 4;
  const bf16_t* p = sbase + colrow * LDS_STRIDE + k0;
  FragB f;
  f.q[0] = *(const uint4*)(p + half * 8);
  f.q[1] = *(const uint4*)(p + 16 + half * 8);
  return f.v;
}

// Async DMA of a 64-row x 64-col bf16 tile (128B/row) into a padded LDS tile.
// 512 16-byte chunks spread over 256 threads (2 per thread), tracked by ASYNCcnt.
__device__ inline void async_tile_load(const bf16_t* __restrict__ gbase, int row0,
                                       int ld, unsigned lds_base, int tid) {
#pragma unroll
  for (int c = 0; c < 2; ++c) {
    const int idx = tid + c * 256;
    const int row = idx >> 3;      // 0..63
    const int col = idx & 7;       // 16B chunk within the 128B row
    const unsigned ldsa = lds_base + (unsigned)(row * (LDS_STRIDE * 2) + col * 16);
    const unsigned long long ga =
        (unsigned long long)(uintptr_t)(gbase + (size_t)(row0 + row) * ld + col * 8);
    asm volatile("global_load_async_to_lds_b128 %0, %1, off"
                 :: "v"(ldsa), "v"(ga) : "memory");
  }
}

// Tensor Data Mover: one D# descriptor moves the whole 64x64 bf16 tile
// (rows of 128B strided by 1024B in global) into the padded LDS tile.
// Hardware padding: pad_interval code 4 (32 DWORDs = 128B) + pad_amount code 3
// (4 DWORDs = 16B) reproduces the 144B LDS row stride. Tracked by TENSORcnt.
__device__ inline void tdm_tile_load(const bf16_t* __restrict__ gtile,
                                     unsigned lds_base) {
  const unsigned long long ga = (unsigned long long)(uintptr_t)gtile;
  v4u g0;
  g0.x = 1u;                                   // count=1 (valid), user mode
  g0.y = lds_base;                             // lds_addr [63:32]
  g0.z = (unsigned)(ga & 0xFFFFFFFFu);         // global_addr lo
  g0.w = (unsigned)((ga >> 32) & 0x01FFFFFFu)  // global_addr hi (57-bit)
         | (2u << 30);                         // type=2 ("image")
  v8i g1;
  g1[0] = (int)((1u << 16)                     // data_size = 2 bytes
                | (1u << 20)                   // pad_enable
                | (4u << 22)                   // pad_interval: 32 DWORDs
                | (3u << 25));                 // pad_amount: 4 DWORDs
  g1[1] = (int)(512u << 16);                   // tensor_dim0 = 512 elems
  g1[2] = (int)(2048u << 16);                  // tensor_dim1 = 2048 rows
  g1[3] = (int)(64u << 16);                    // tile_dim0 = 64
  g1[4] = 64;                                  // tile_dim1 = 64
  g1[5] = 512;                                 // tensor_dim0_stride = 512 elems
  g1[6] = 0;
  g1[7] = 0;
  v4i gz  = {0, 0, 0, 0};                      // groups 2/3 unused (2D tensor)
  v8i gz8 = {0, 0, 0, 0, 0, 0, 0, 0};          // trailing group (clang-23 form)
  // amdgpu-toolchain (clang-23) 6-arg form:
  // (uint32x4 g0, int32x8 g1, int32x4, int32x4, int32x8, i32 cpol)
  __builtin_amdgcn_tensor_load_to_lds(g0, g1, gz, gz, gz8, 0);
}

// ---------------------------------------------------------------------------
// Kernel 1: fp32 -> bf16 conversions (x + four weight matrices)
// ---------------------------------------------------------------------------
__global__ void k_convert_inputs(const float* __restrict__ x,
                                 const float* __restrict__ wq,
                                 const float* __restrict__ wk,
                                 const float* __restrict__ wv,
                                 const float* __restrict__ wo,
                                 bf16_t* __restrict__ xb,
                                 bf16_t* __restrict__ wqb,
                                 bf16_t* __restrict__ wkb,
                                 bf16_t* __restrict__ wvb,
                                 bf16_t* __restrict__ wob) {
  int i = blockIdx.x * blockDim.x + threadIdx.x;
  const int ND = T_SEQ * D_MODEL;
  const int NW = D_MODEL * D_MODEL;
  if (i < ND) xb[i] = f32_to_bf16(x[i]);
  if (i < NW) {
    wqb[i] = f32_to_bf16(wq[i]);
    wkb[i] = f32_to_bf16(wk[i]);
    wvb[i] = f32_to_bf16(wv[i]);
    wob[i] = f32_to_bf16(wo[i]);
  }
}

// ---------------------------------------------------------------------------
// Kernel 2: fused QKV projection, y = x @ W^T + b  (one wave = one 16x16 tile,
// K-loop pipelines global b128 loads against WMMA; operands re-hit 192MB L2)
// ---------------------------------------------------------------------------
__global__ void __launch_bounds__(32)
k_gemm_qkv(const bf16_t* __restrict__ xb,
           const bf16_t* __restrict__ wqb, const bf16_t* __restrict__ wkb,
           const bf16_t* __restrict__ wvb,
           const float* __restrict__ bq, const float* __restrict__ bk,
           const float* __restrict__ bv,
           float* __restrict__ qf, float* __restrict__ kf, float* __restrict__ vf,
           bf16_t* __restrict__ qb, bf16_t* __restrict__ kb) {
  const int lane = threadIdx.x & 31;
  const int m0 = blockIdx.x * 16;
  const int n0 = blockIdx.y * 16;
  const int which = blockIdx.z;
  const bf16_t* wb   = (which == 0) ? wqb : (which == 1) ? wkb : wvb;
  const float*  bias = (which == 0) ? bq  : (which == 1) ? bk  : bv;
  float*  outf = (which == 0) ? qf : (which == 1) ? kf : vf;
  bf16_t* outb = (which == 0) ? qb : (which == 1) ? kb : nullptr;

  const int r = lane & 15;
  const int half = lane >> 4;
  __builtin_prefetch(xb + (size_t)(m0 + r) * D_MODEL, 0, 1); // global_prefetch_b8
  v8f c = {};
  for (int k0 = 0; k0 < D_MODEL; k0 += 32) {
    v16bf a = load_frag_a(xb, m0 + r, D_MODEL, k0, lane);
    v16bf b = load_frag_b(wb, n0 + r, D_MODEL, k0, lane); // W row n = B column n
    c = __builtin_amdgcn_wmma_f32_16x16x32_bf16(false, a, false, b,
                                                (short)0, c, false, false);
  }
  const float bn = bias[n0 + r];
  for (int j = 0; j < 8; ++j) {
    const int row = m0 + j + half * 8;
    const float val = c[j] + bn;
    outf[(size_t)row * D_MODEL + n0 + r] = val;
    if (outb) outb[(size_t)row * D_MODEL + n0 + r] = f32_to_bf16(val);
  }
}

// ---------------------------------------------------------------------------
// Kernel 3: per-head scores = (Q_h @ K_h^T) / sqrt(hd)   [H, T, T] f32
// Workgroup = 256 threads (8 waves) producing a 64x64 output tile.
// Q tile: TENSOR_LOAD_TO_LDS (TDM descriptor, hardware row padding).
// K tile: GLOBAL_LOAD_ASYNC_TO_LDS_B128 (per-lane async DMA, ASYNCcnt).
// Fragments then read out of LDS -> 4x operand reuse per fetch.
// ---------------------------------------------------------------------------
__global__ void __launch_bounds__(256)
k_scores(const bf16_t* __restrict__ qb, const bf16_t* __restrict__ kb,
         float* __restrict__ scores) {
  __shared__ bf16_t sQ[64 * LDS_STRIDE];
  __shared__ bf16_t sK[64 * LDS_STRIDE];
  const int tid  = threadIdx.x;
  const int wave = tid >> 5;
  const int lane = tid & 31;
  const int r = lane & 15, half = lane >> 4;
  const int t0 = blockIdx.x * 64;   // query tile base
  const int s0 = blockIdx.y * 64;   // key tile base
  const int h  = blockIdx.z;
  const bf16_t* qh = qb + h * HD;
  const bf16_t* kh = kb + h * HD;

  const unsigned q_lds = (unsigned)(uintptr_t)(&sQ[0]);
  const unsigned k_lds = (unsigned)(uintptr_t)(&sK[0]);
  // K tile: fire-and-forget per-lane async DMA
  async_tile_load(kh, s0, D_MODEL, k_lds, tid);
  // Q tile: one TDM descriptor from wave 0 (TDM ignores EXEC; once is enough)
  if (wave == 0) {
    tdm_tile_load(qh + (size_t)t0 * D_MODEL, q_lds);
    __builtin_amdgcn_s_wait_tensorcnt(0);     // drain TENSORcnt
  }
  asm volatile("s_wait_asynccnt 0" ::: "memory"); // drain this wave's ASYNCcnt
  __syncthreads();                                // tiles visible WGP-wide

  const float scale = 0.125f; // 1/sqrt(64)
  float* out = scores + (size_t)h * T_SEQ * T_SEQ;
  const int mi = wave & 3;          // 4x4 grid of 16x16 tiles, 2 per wave
#pragma unroll
  for (int nt = 0; nt < 2; ++nt) {
    const int ni = ((wave >> 2) << 1) + nt;
    v8f c = {};
#pragma unroll
    for (int k0 = 0; k0 < HD; k0 += 32) {
      v16bf a = lds_frag_a(sQ, mi * 16 + r, k0, lane);
      v16bf b = lds_frag_b(sK, ni * 16 + r, k0, lane);
      c = __builtin_amdgcn_wmma_f32_16x16x32_bf16(false, a, false, b,
                                                  (short)0, c, false, false);
    }
#pragma unroll
    for (int j = 0; j < 8; ++j)
      out[(size_t)(t0 + mi * 16 + j + half * 8) * T_SEQ + s0 + ni * 16 + r] =
          c[j] * scale;
  }
}

// ---------------------------------------------------------------------------
// Kernel 4: per-row quantile thresholds via value-space binary search in LDS
// thr[row] = {crit_th, skip_th, marginal_fraction}
// ---------------------------------------------------------------------------
__global__ void __launch_bounds__(128)
k_thresholds(const float* __restrict__ scores, float* __restrict__ thr) {
  __shared__ float srow[T_SEQ];
  __shared__ float redf[128];
  __shared__ int   cnt;
  const int row = blockIdx.x;            // h*T + t
  const int tid = threadIdx.x;
  const float* s = scores + (size_t)row * T_SEQ;
  float lmin = 1e30f, lmax = -1e30f;
  for (int i = tid; i < T_SEQ; i += 128) {
    float v = s[i];
    srow[i] = v;
    lmin = fminf(lmin, v);
    lmax = fmaxf(lmax, v);
  }
  redf[tid] = lmax; __syncthreads();
  for (int st = 64; st > 0; st >>= 1) {
    if (tid < st) redf[tid] = fmaxf(redf[tid], redf[tid + st]);
    __syncthreads();
  }
  const float vmax = redf[0]; __syncthreads();
  redf[tid] = lmin; __syncthreads();
  for (int st = 64; st > 0; st >>= 1) {
    if (tid < st) redf[tid] = fminf(redf[tid], redf[tid + st]);
    __syncthreads();
  }
  const float vmin = redf[0]; __syncthreads();

  auto select_q = [&](int target) -> float {
    float lo = vmin, hi = vmax;
    for (int it = 0; it < 24; ++it) {
      float mid = 0.5f * (lo + hi);
      if (tid == 0) cnt = 0;
      __syncthreads();
      int lc = 0;
      for (int i = tid; i < T_SEQ; i += 128) lc += (srow[i] <= mid) ? 1 : 0;
      atomicAdd(&cnt, lc);
      __syncthreads();
      int c = cnt;
      __syncthreads();
      if (c >= target) hi = mid; else lo = mid;
    }
    return hi;
  };
  const int tgt_crit = (int)((1.0f - K_CRIT) * (T_SEQ - 1)) + 1; // ~rank 1945
  const int tgt_skip = (int)(K_SKIP * (T_SEQ - 1)) + 1;          // ~rank 205
  const float crit_th = select_q(tgt_crit);
  const float skip_th = select_q(tgt_skip);

  if (tid == 0) cnt = 0;
  __syncthreads();
  int lc = 0;
  for (int i = tid; i < T_SEQ; i += 128) {
    float v = srow[i];
    lc += (v >= skip_th && v <= crit_th) ? 1 : 0;
  }
  atomicAdd(&cnt, lc);
  __syncthreads();
  if (tid == 0) {
    thr[(size_t)row * 3 + 0] = crit_th;
    thr[(size_t)row * 3 + 1] = skip_th;
    thr[(size_t)row * 3 + 2] = (float)cnt / (float)T_SEQ;
  }
}

// ---------------------------------------------------------------------------
// Kernel 5: critical path — masked softmax over scores > crit_th, then attn@V
// Writes comb[t, h*64+d]  (overwrites; linear path adds afterwards)
// ---------------------------------------------------------------------------
__global__ void __launch_bounds__(256)
k_crit_out(const float* __restrict__ scores, const float* __restrict__ thr,
           const float* __restrict__ vf, float* __restrict__ comb) {
  __shared__ float redf[256];
  __shared__ float wbuf[256];
  __shared__ float accP[256];
  const int t = blockIdx.x, h = blockIdx.y, tid = threadIdx.x;
  const size_t rowi = (size_t)h * T_SEQ + t;
  const float* s = scores + rowi * T_SEQ;
  const float crit_th = thr[rowi * 3 + 0];
  // pass 1: masked max
  float lm = -1e30f;
  for (int i = tid; i < T_SEQ; i += 256) {
    float v = s[i];
    if (v > crit_th) lm = fmaxf(lm, v);
  }
  redf[tid] = lm; __syncthreads();
  for (int st = 128; st > 0; st >>= 1) {
    if (tid < st) redf[tid] = fmaxf(redf[tid], redf[tid + st]);
    __syncthreads();
  }
  const float m = redf[0]; __syncthreads();
  // pass 2: partition function
  float lz = 0.f;
  for (int i = tid; i < T_SEQ; i += 256) {
    float v = s[i];
    if (v > crit_th) lz += __expf(v - m);
  }
  redf[tid] = lz; __syncthreads();
  for (int st = 128; st > 0; st >>= 1) {
    if (tid < st) redf[tid] += redf[tid + st];
    __syncthreads();
  }
  const float Z = fmaxf(redf[0], 1e-20f); __syncthreads();
  // pass 3: weighted accumulation of V rows — all 256 threads active:
  // thread (q,d) with d=tid&63, q=tid>>6 accumulates quarter q of each chunk.
  const int d = tid & 63, qq = tid >> 6;
  float acc = 0.f;
  for (int jb = 0; jb < T_SEQ; jb += 256) {
    float v = s[jb + tid];
    wbuf[tid] = (v > crit_th) ? __expf(v - m) : 0.f;
    __syncthreads();
    const float* vr = vf + (size_t)(jb + qq * 64) * D_MODEL + h * HD + d;
#pragma unroll 4
    for (int jj = 0; jj < 64; ++jj)
      acc += wbuf[qq * 64 + jj] * vr[(size_t)jj * D_MODEL];
    __syncthreads();
  }
  accP[tid] = acc;
  __syncthreads();
  if (tid < HD) {
    float a = accP[d] + accP[64 + d] + accP[128 + d] + accP[192 + d];
    comb[(size_t)t * D_MODEL + h * HD + d] = a / Z;
  }
}

// ---------------------------------------------------------------------------
// Kernel 6: causal linear attention — per-head running KV state in LDS.
// phi(x) = elu(x)+1. Inclusive update, then (q·S)/(q·z) * marginal_fraction.
// ---------------------------------------------------------------------------
__global__ void __launch_bounds__(256)
k_linear(const float* __restrict__ qf, const float* __restrict__ kf,
         const float* __restrict__ vf, const float* __restrict__ thr,
         float* __restrict__ comb) {
  __shared__ float S[HD * (HD + 1)];  // stride 65 -> conflict-free column reads
  __shared__ float zacc[HD];
  __shared__ float qs[HD], ks[HD], vs[HD];
  __shared__ float numP[256];
  __shared__ float red[64];
  const int h = blockIdx.x, tid = threadIdx.x;
  for (int i = tid; i < HD * (HD + 1); i += 256) S[i] = 0.f;
  if (tid < HD) zacc[tid] = 0.f;
  __syncthreads();
  const int d = tid & 63, p = tid >> 6;
  for (int t = 0; t < T_SEQ; ++t) {
    if (tid < HD) {
      const size_t idx = (size_t)t * D_MODEL + h * HD + tid;
      const float qv = qf[idx], kv = kf[idx];
      qs[tid] = (qv > 0.f) ? qv + 1.f : __expf(qv);
      ks[tid] = (kv > 0.f) ? kv + 1.f : __expf(kv);
      vs[tid] = vf[idx];
    }
    __syncthreads();
    // S += ks ⊗ vs (16 entries per thread); z += ks
    {
      const int base = tid * 16;
      for (int i = 0; i < 16; ++i) {
        const int idx = base + i;
        const int rr = idx >> 6, cc = idx & 63;
        S[rr * (HD + 1) + cc] += ks[rr] * vs[cc];
      }
      if (tid < HD) zacc[tid] += ks[tid];
    }
    __syncthreads();
    // numerator partials: 4 partial sums per output dim
    float part = 0.f;
    for (int rr = p * 16; rr < p * 16 + 16; ++rr)
      part += qs[rr] * S[rr * (HD + 1) + d];
    numP[tid] = part;
    if (tid < HD) red[tid] = qs[tid] * zacc[tid];
    __syncthreads();
    for (int st = 32; st > 0; st >>= 1) {
      if (tid < st) red[tid] += red[tid + st];
      __syncthreads();
    }
    if (tid < HD) {
      const float den = fmaxf(red[0], 1e-6f);
      const float num = numP[d] + numP[64 + d] + numP[128 + d] + numP[192 + d];
      const float mfrac = thr[((size_t)h * T_SEQ + t) * 3 + 2];
      comb[(size_t)t * D_MODEL + h * HD + tid] += (num / den) * mfrac;
    }
    __syncthreads();
  }
}

// ---------------------------------------------------------------------------
// Kernel 7a: comb fp32 -> bf16 for the output projection
// ---------------------------------------------------------------------------
__global__ void k_convert_comb(const float* __restrict__ comb,
                               bf16_t* __restrict__ combb) {
  int i = blockIdx.x * blockDim.x + threadIdx.x;
  combb[i] = f32_to_bf16(comb[i]);
}

// ---------------------------------------------------------------------------
// Kernel 7b: output projection out = comb @ Wo^T + bo (WMMA bf16)
// ---------------------------------------------------------------------------
__global__ void __launch_bounds__(32)
k_out_proj(const bf16_t* __restrict__ combb, const bf16_t* __restrict__ wob,
           const float* __restrict__ bo, float* __restrict__ out) {
  const int lane = threadIdx.x & 31;
  const int m0 = blockIdx.x * 16;
  const int n0 = blockIdx.y * 16;
  const int r = lane & 15, half = lane >> 4;
  v8f c = {};
  for (int k0 = 0; k0 < D_MODEL; k0 += 32) {
    v16bf a = load_frag_a(combb, m0 + r, D_MODEL, k0, lane);
    v16bf b = load_frag_b(wob, n0 + r, D_MODEL, k0, lane);
    c = __builtin_amdgcn_wmma_f32_16x16x32_bf16(false, a, false, b,
                                                (short)0, c, false, false);
  }
  const float bn = bo[n0 + r];
  for (int j = 0; j < 8; ++j)
    out[(size_t)(m0 + j + half * 8) * D_MODEL + n0 + r] = c[j] + bn;
}

// ---------------------------------------------------------------------------
// Host launcher
// ---------------------------------------------------------------------------
extern "C" void kernel_launch(void* const* d_in, const int* in_sizes, int n_in,
                              void* d_out, int out_size, void* d_ws, size_t ws_size,
                              hipStream_t stream) {
  (void)in_sizes; (void)n_in; (void)out_size; (void)ws_size;
  const float* x  = (const float*)d_in[0];
  const float* wq = (const float*)d_in[1];
  const float* bq = (const float*)d_in[2];
  const float* wk = (const float*)d_in[3];
  const float* bk = (const float*)d_in[4];
  const float* wv = (const float*)d_in[5];
  const float* bv = (const float*)d_in[6];
  const float* wo = (const float*)d_in[7];
  const float* bo = (const float*)d_in[8];

  char* ws = (char*)d_ws;
  size_t off = 0;
  auto alloc = [&](size_t bytes) -> char* {
    char* p = ws + off;
    off += (bytes + 255) & ~(size_t)255;
    return p;
  };
  const size_t ND = (size_t)T_SEQ * D_MODEL;
  const size_t NW = (size_t)D_MODEL * D_MODEL;
  bf16_t* xb     = (bf16_t*)alloc(ND * 2);
  bf16_t* wqb    = (bf16_t*)alloc(NW * 2);
  bf16_t* wkb    = (bf16_t*)alloc(NW * 2);
  bf16_t* wvb    = (bf16_t*)alloc(NW * 2);
  bf16_t* wob    = (bf16_t*)alloc(NW * 2);
  float*  qf     = (float*)alloc(ND * 4);
  float*  kf     = (float*)alloc(ND * 4);
  float*  vf     = (float*)alloc(ND * 4);
  bf16_t* qb     = (bf16_t*)alloc(ND * 2);
  bf16_t* kb     = (bf16_t*)alloc(ND * 2);
  float*  scores = (float*)alloc((size_t)H_HEADS * T_SEQ * T_SEQ * 4); // 128 MB, fits L2
  float*  thr    = (float*)alloc((size_t)H_HEADS * T_SEQ * 3 * 4);
  float*  comb   = (float*)alloc(ND * 4);
  bf16_t* combb  = (bf16_t*)alloc(ND * 2);
  float*  out    = (float*)d_out;

  // 1. conversions
  k_convert_inputs<<<dim3((unsigned)(ND + 255) / 256), 256, 0, stream>>>(
      x, wq, wk, wv, wo, xb, wqb, wkb, wvb, wob);
  // 2. QKV projections (z = which of q/k/v)
  k_gemm_qkv<<<dim3(T_SEQ / 16, D_MODEL / 16, 3), 32, 0, stream>>>(
      xb, wqb, wkb, wvb, bq, bk, bv, qf, kf, vf, qb, kb);
  // 3. per-head attention scores (64x64 tile per workgroup, TDM + async-to-LDS)
  k_scores<<<dim3(T_SEQ / 64, T_SEQ / 64, H_HEADS), 256, 0, stream>>>(qb, kb, scores);
  // 4. per-row quantile thresholds
  k_thresholds<<<dim3(H_HEADS * T_SEQ), 128, 0, stream>>>(scores, thr);
  // 5. critical softmax path
  k_crit_out<<<dim3(T_SEQ, H_HEADS), 256, 0, stream>>>(scores, thr, vf, comb);
  // 6. causal linear attention path (adds into comb)
  k_linear<<<dim3(H_HEADS), 256, 0, stream>>>(qf, kf, vf, thr, comb);
  // 7. output projection
  k_convert_comb<<<dim3((unsigned)(ND + 255) / 256), 256, 0, stream>>>(comb, combb);
  k_out_proj<<<dim3(T_SEQ / 16, D_MODEL / 16), 32, 0, stream>>>(combb, wob, bo, out);
}
